// DecoderRNN_77988016161287
// MI455X (gfx1250) — matrix operand order repaired and checked by reading, not compiled, not run on hardware
//
#include <hip/hip_runtime.h>
#include <hip/hip_bf16.h>

#define HIDDEN 2048
#define GATES  8192
#define VOCAB  7
#define STEPS  1024
#define NWG    128          // one 16-hidden-index slice per workgroup
#define WGSIZE 256          // 8 waves: 4 gate types x 2 K-halves
#define FRAGS  32           // 32 x K32 fragments = 1024 K per wave

typedef __attribute__((ext_vector_type(16))) __bf16   v16bf;
typedef __attribute__((ext_vector_type(8)))  float    v8f;
typedef __attribute__((ext_vector_type(4)))  unsigned v4u;

union Frag { v16bf v; v4u u[2]; };

// ---- workspace layout (bytes) ----
#define WS_WSW   0u                      // swizzled bf16 W_hh: 1024 waves * 16384 ushort = 32 MB
#define WS_BIAS  33554432u               // biasTab[v][g] = W_ih[g][v]+b_ih[g]+b_hh[g] : 7*8192 f32
#define WS_HBF   (WS_BIAS + 229376u)     // h in bf16: 2048 ushort
#define WS_LOG   (WS_HBF + 4096u)        // raw logits: 1024*7 f32
#define WS_BAR   (WS_LOG + 28672u)       // barrier block: 512 ints (8 L0 + L1 + gen, 128B apart)
// barrier int indices (within WS_BAR, 128B = 32-int spacing to avoid line sharing)
#define BAR_L0(g) ((g) * 32)
#define BAR_L1    256
#define BAR_GEN   288

__device__ __forceinline__ unsigned short f2bf(float f) {
    unsigned x = __float_as_uint(f);
    unsigned r = x + 0x7FFFu + ((x >> 16) & 1u);   // round-to-nearest-even
    return (unsigned short)(r >> 16);
}

// One-shot setup: swizzle W_hh into per-wave WMMA-B fragment order (bf16),
// fold W_ih columns + both biases into a token-indexed table, seed h (bf16),
// seed raw logits with b_out, reset the grid barrier.
__global__ void init_kernel(const float* __restrict__ h0,
                            const float* __restrict__ W_ih,
                            const float* __restrict__ W_hh,
                            const float* __restrict__ b_ih,
                            const float* __restrict__ b_hh,
                            const float* __restrict__ b_out,
                            unsigned char* __restrict__ ws) {
    size_t id = (size_t)blockIdx.x * blockDim.x + threadIdx.x;

    // B-fragment swizzle: [wave][frag][lane][slot] ; 16,777,216 elements
    if (id < (size_t)16777216) {
        unsigned s    = (unsigned)(id & 15u);        // bf16 slot within lane (K offset)
        unsigned l    = (unsigned)((id >> 4) & 31u); // lane
        unsigned f    = (unsigned)((id >> 9) & 31u); // fragment
        unsigned wave = (unsigned)(id >> 14);        // 0..1023
        unsigned wg = wave >> 3, w = wave & 7;
        unsigned gtype = w & 3, khalf = w >> 2;
        unsigned n = l & 15, half = l >> 4;
        unsigned gate = gtype * 2048u + wg * 16u + n;          // N column -> gate row
        unsigned K    = khalf * 1024u + f * 32u + (s >> 3) * 16u + half * 8u + (s & 7u);
        ((unsigned short*)(ws + WS_WSW))[id] = f2bf(W_hh[(size_t)gate * HIDDEN + K]);
    }
    if (id < (size_t)VOCAB * GATES) {  // biasTab
        unsigned v = (unsigned)(id / GATES), g = (unsigned)(id % GATES);
        ((float*)(ws + WS_BIAS))[id] = W_ih[(size_t)g * VOCAB + v] + b_ih[g] + b_hh[g];
    }
    if (id < HIDDEN)            ((unsigned short*)(ws + WS_HBF))[id] = f2bf(h0[id]);
    if (id < STEPS * VOCAB)     ((float*)(ws + WS_LOG))[id] = b_out[id % VOCAB];
    if (id < 512)               ((int*)(ws + WS_BAR))[id] = 0;
}

// Persistent kernel: weights live in VGPRs for all 1024 steps.
__global__ __launch_bounds__(WGSIZE, 1)
void lstm_persistent(const float* __restrict__ c0,
                     const float* __restrict__ W_out,
                     unsigned char* __restrict__ ws,
                     float* __restrict__ d_out) {
    // single struct so `h` is at LDS offset 0 (async-to-LDS dst operand = byte offset)
    __shared__ struct {
        unsigned short h[HIDDEN];   // 4 KB broadcast copy of h (bf16), offset 0
        float gpart[(WGSIZE / 32) * 16];
        float c[16];
        float h2[16];
        int   idx;
    } S;

    const int tid   = threadIdx.x;
    const int wg    = blockIdx.x;
    const int w     = tid >> 5;          // wave in WG
    const int lane  = tid & 31;
    const int khalf = w >> 2;            // which 1024-wide K half
    const int half  = lane >> 4;         // lane half (A-layout K interleave)

    unsigned short* Wsw     = (unsigned short*)(ws + WS_WSW);
    const float*    biasTab = (const float*)(ws + WS_BIAS);
    unsigned short* hbf     = (unsigned short*)(ws + WS_HBF);
    float*          lraw    = (float*)(ws + WS_LOG);
    int*            bar     = (int*)(ws + WS_BAR);

    // ---- load this wave's 16x1024 bf16 weight slab into 256 VGPRs, once ----
    Frag b[FRAGS];
    {
        const v4u* src = (const v4u*)(Wsw + (size_t)(wg * 8 + w) * 16384);
#pragma unroll
        for (int f = 0; f < FRAGS; ++f) {
            b[f].u[0] = src[f * 64 + lane * 2 + 0];
            b[f].u[1] = src[f * 64 + lane * 2 + 1];
        }
    }
    if (tid < 16) S.c[tid] = c0[wg * 16 + tid];

    // register-resident activation constants (avoid per-step L2/L0 loads on the
    // serial path): W_out slice for the logit partial, biasTab for all 7 tokens
    float wo[16];
    if (tid < VOCAB) {
        const float* p = W_out + (size_t)tid * HIDDEN + wg * 16;
#pragma unroll
        for (int j = 0; j < 16; ++j) wo[j] = p[j];
    }
    float btr[VOCAB][4];
    if (tid < 16) {
#pragma unroll
        for (int v = 0; v < VOCAB; ++v)
#pragma unroll
            for (int g = 0; g < 4; ++g)
                btr[v][g] = biasTab[(size_t)v * GATES + g * 2048 + wg * 16 + tid];
    }

    for (int t = 0; t < STEPS; ++t) {
        // ---- async direct-to-LDS broadcast of h (256 lanes x 16B = 4 KB) ----
        {
            unsigned ldsoff = (unsigned)(tid * 16);
            unsigned long long gaddr =
                (unsigned long long)(uintptr_t)(hbf + (size_t)tid * 8);
            asm volatile("global_load_async_to_lds_b128 %0, %1, off scope:SCOPE_DEV"
                         :: "v"(ldsoff), "v"(gaddr) : "memory");
            asm volatile("s_wait_asynccnt 0x0" ::: "memory");
        }
        // issue last-step logit loads early; consume AFTER the WMMA loop so the
        // L2 round-trip hides under matrix work
        float lg[VOCAB];
        if (tid == 0) {
            const float* lr = lraw + (size_t)(t ? t - 1 : 0) * VOCAB;
#pragma unroll
            for (int v = 0; v < VOCAB; ++v) lg[v] = lr[v];
        }
        __syncthreads();

        // ---- GEMV core: 32 chained v_wmma_f32_16x16x32_bf16 ----
        // sched_group_barrier pins the software pipeline: loads for fragment
        // f+3 issue before WMMA f, so LDS latency hides under 3 WMMAs.
        v8f acc = {};
        const char* hb = (const char*)&S.h[0] + khalf * 2048 + half * 16;
        Frag a[4];
#pragma unroll
        for (int p = 0; p < 3; ++p) {
            a[p].u[0] = *(const v4u*)(hb + p * 64);
            a[p].u[1] = *(const v4u*)(hb + p * 64 + 32);
        }
        __builtin_amdgcn_sched_group_barrier(0x100, 6, 0);   // 6 preamble DS reads first
#pragma unroll
        for (int f = 0; f < FRAGS; ++f) {
            if (f + 3 < FRAGS) {
                a[(f + 3) & 3].u[0] = *(const v4u*)(hb + (f + 3) * 64);
                a[(f + 3) & 3].u[1] = *(const v4u*)(hb + (f + 3) * 64 + 32);
            }
            acc = __builtin_amdgcn_wmma_f32_16x16x32_bf16(false, a[f & 3].v, false, b[f].v,
                                                          (short)0, acc, false, false);
            __builtin_amdgcn_sched_group_barrier(0x008, 1, 0);       // 1 WMMA
            if (f + 3 < FRAGS)
                __builtin_amdgcn_sched_group_barrier(0x100, 2, 0);   // then 2 DS reads
        }

        // greedy feedback token (loads issued pre-WMMA)
        if (tid == 0) {
            int bi = VOCAB - 1;              // SOS one-hot at t==0
            if (t > 0) {
                float best = lg[0]; bi = 0;
#pragma unroll
                for (int v = 1; v < VOCAB; ++v) { if (lg[v] > best) { best = lg[v]; bi = v; } }
            }
            S.idx = bi;
        }
        // D rows are replicated; lane n<16 holds the full dot for gate column n
        if (lane < 16) S.gpart[w * 16 + lane] = acc[0];
        __syncthreads();

        // ---- LSTM cell update for this WG's 16 hidden indices ----
        if (tid < 16) {
            const int j = tid, hidx = wg * 16 + j;
            const int idx = S.idx;
            float bi = 0.f, bf = 0.f, bg = 0.f, bo = 0.f;
#pragma unroll
            for (int v = 0; v < VOCAB; ++v) {     // register select, no loads
                bool m = (idx == v);
                bi = m ? btr[v][0] : bi;
                bf = m ? btr[v][1] : bf;
                bg = m ? btr[v][2] : bg;
                bo = m ? btr[v][3] : bo;
            }
            float gi = S.gpart[0 * 16 + j] + S.gpart[4 * 16 + j] + bi;
            float gf = S.gpart[1 * 16 + j] + S.gpart[5 * 16 + j] + bf;
            float gg = S.gpart[2 * 16 + j] + S.gpart[6 * 16 + j] + bg;
            float go = S.gpart[3 * 16 + j] + S.gpart[7 * 16 + j] + bo;
            float si = 1.f / (1.f + __expf(-gi));
            float sf = 1.f / (1.f + __expf(-gf));
            float so = 1.f / (1.f + __expf(-go));
            float c2 = sf * S.c[j] + si * tanhf(gg);
            float h2 = so * tanhf(c2);
            S.c[j]  = c2;
            S.h2[j] = h2;
            hbf[hidx] = f2bf(h2);
            if (t == STEPS - 1) {            // tuple outputs hN, cN
                d_out[STEPS * VOCAB + hidx]          = h2;
                d_out[STEPS * VOCAB + HIDDEN + hidx] = c2;
            }
        }
        __syncthreads();

        // partial output head: 7 x 16 MACs from registers, one atomic per vocab entry
        if (tid < VOCAB) {
            float p = 0.f;
#pragma unroll
            for (int j = 0; j < 16; ++j) p += wo[j] * S.h2[j];
            atomicAdd(&lraw[(size_t)t * VOCAB + tid], p);
        }
        __threadfence();

        // ---- two-level grid barrier: 8 groups of 16, then 8-wide fan-in ----
        __syncthreads();
        if (tid == 0) {
            int gen = __hip_atomic_load(&bar[BAR_GEN], __ATOMIC_ACQUIRE, __HIP_MEMORY_SCOPE_AGENT);
            bool winner = false;
            int grp = wg >> 4;
            int n0 = __hip_atomic_fetch_add(&bar[BAR_L0(grp)], 1, __ATOMIC_ACQ_REL, __HIP_MEMORY_SCOPE_AGENT);
            if (n0 == 15) {
                __hip_atomic_store(&bar[BAR_L0(grp)], 0, __ATOMIC_RELAXED, __HIP_MEMORY_SCOPE_AGENT);
                int n1 = __hip_atomic_fetch_add(&bar[BAR_L1], 1, __ATOMIC_ACQ_REL, __HIP_MEMORY_SCOPE_AGENT);
                if (n1 == 7) {
                    __hip_atomic_store(&bar[BAR_L1], 0, __ATOMIC_RELAXED, __HIP_MEMORY_SCOPE_AGENT);
                    __hip_atomic_fetch_add(&bar[BAR_GEN], 1, __ATOMIC_RELEASE, __HIP_MEMORY_SCOPE_AGENT);
                    winner = true;
                }
            }
            if (!winner) {
                while (__hip_atomic_load(&bar[BAR_GEN], __ATOMIC_ACQUIRE, __HIP_MEMORY_SCOPE_AGENT) == gen)
                    __builtin_amdgcn_s_sleep(2);
            }
        }
        __syncthreads();
    }
}

__global__ void logsoftmax_kernel(const unsigned char* __restrict__ ws,
                                  float* __restrict__ out) {
    int t = blockIdx.x * blockDim.x + threadIdx.x;
    if (t >= STEPS) return;
    const float* lr = (const float*)(ws + WS_LOG) + (size_t)t * VOCAB;
    float m = lr[0];
    for (int v = 1; v < VOCAB; ++v) m = fmaxf(m, lr[v]);
    float s = 0.f;
    for (int v = 0; v < VOCAB; ++v) s += __expf(lr[v] - m);
    float lse = __logf(s);
    for (int v = 0; v < VOCAB; ++v) out[(size_t)t * VOCAB + v] = lr[v] - m - lse;
}

extern "C" void kernel_launch(void* const* d_in, const int* in_sizes, int n_in,
                              void* d_out, int out_size, void* d_ws, size_t ws_size,
                              hipStream_t stream) {
    (void)in_sizes; (void)n_in; (void)out_size; (void)ws_size;
    const float* h0    = (const float*)d_in[0];
    const float* c0    = (const float*)d_in[1];
    const float* W_ih  = (const float*)d_in[2];
    const float* W_hh  = (const float*)d_in[3];
    const float* b_ih  = (const float*)d_in[4];
    const float* b_hh  = (const float*)d_in[5];
    const float* W_out = (const float*)d_in[6];
    const float* b_out = (const float*)d_in[7];
    unsigned char* ws  = (unsigned char*)d_ws;
    float* out         = (float*)d_out;

    init_kernel<<<65536, 256, 0, stream>>>(h0, W_ih, W_hh, b_ih, b_hh, b_out, ws);
    lstm_persistent<<<NWG, WGSIZE, 0, stream>>>(c0, W_out, ws, out);
    logsoftmax_kernel<<<(STEPS + 255) / 256, 256, 0, stream>>>(ws, out);
}
